// UnsuperNet_28535762715332
// MI455X (gfx1250) — matrix operand order repaired and checked by reading, not compiled, not run on hardware
//
#include <hip/hip_runtime.h>
#include <stdint.h>

// ---------------------------------------------------------------------------
// Types for CDNA5 WMMA (wave32, 16x16x32 bf16 -> f32)
// ---------------------------------------------------------------------------
typedef __attribute__((ext_vector_type(16))) __bf16    bf16x16;
typedef __attribute__((ext_vector_type(8)))  float     f32x8;
typedef __attribute__((ext_vector_type(4)))  uint32_t  u32x4;

union ABPack { u32x4 q[2]; bf16x16 v; };   // 32 bytes, trivially constructible

#define DEVFN __device__ __forceinline__

DEVFN uint16_t f32_to_bf16(float f) {
  uint32_t x = __float_as_uint(f);
  uint32_t r = x + 0x7FFFu + ((x >> 16) & 1u);   // round-to-nearest-even
  return (uint16_t)(r >> 16);
}
DEVFN float bf16_to_f32(uint16_t h) { return __uint_as_float(((uint32_t)h) << 16); }

DEVFN f32x8 wmma_bf16(const ABPack& a, const ABPack& b, f32x8 c) {
  return __builtin_amdgcn_wmma_f32_16x16x32_bf16(
      false, a.v, false, b.v, (short)0, c, false, false);
}

// ---------------------------------------------------------------------------
// Prep: fold BN into conv weights (bf16, K-major k = (ky*3+kx)*Cin + ci)
// ---------------------------------------------------------------------------
__global__ void fold_weights(const float* __restrict__ w, const float* __restrict__ b,
                             const float* __restrict__ gamma, const float* __restrict__ beta,
                             const float* __restrict__ mean, const float* __restrict__ var,
                             uint16_t* __restrict__ wb, float* __restrict__ bias,
                             int Cin, int Cout) {
  int i = blockIdx.x * blockDim.x + threadIdx.x;
  int K = Cin * 9;
  if (i < Cout)
    bias[i] = (b[i] - mean[i]) * gamma[i] * rsqrtf(var[i] + 1e-5f) + beta[i];
  if (i >= Cout * K) return;
  int co = i / K, k = i - co * K;
  int s  = k / Cin, ci = k - s * Cin;
  int ky = s / 3,  kx = s - ky * 3;
  float scale = gamma[co] * rsqrtf(var[co] + 1e-5f);
  wb[i] = f32_to_bf16(w[((co * Cin + ci) * 3 + ky) * 3 + kx] * scale);
}

__global__ void cvt_bf16(const float* __restrict__ in, uint16_t* __restrict__ out, int n) {
  int i = blockIdx.x * blockDim.x + threadIdx.x;
  if (i < n) out[i] = f32_to_bf16(in[i]);
}

// ---------------------------------------------------------------------------
// conv1: Cin=2 (K=18 < 32) -> direct fp32, BN folded inline, bf16 NHWC out
// ---------------------------------------------------------------------------
__global__ void conv1_direct(const float* __restrict__ img,   // [N,2,128,128]
                             const float* __restrict__ w, const float* __restrict__ b,
                             const float* __restrict__ gamma, const float* __restrict__ beta,
                             const float* __restrict__ mean, const float* __restrict__ var,
                             uint16_t* __restrict__ out,      // [N,126,126,64] bf16
                             int N) {
  int idx = blockIdx.x * blockDim.x + threadIdx.x;
  int co = idx & 63;
  int t  = idx >> 6;
  int ox = t % 126; t /= 126;
  int oy = t % 126; t /= 126;
  int n  = t;
  if (n >= N) return;
  float acc = 0.f;
  #pragma unroll
  for (int ci = 0; ci < 2; ++ci)
    #pragma unroll
    for (int ky = 0; ky < 3; ++ky)
      #pragma unroll
      for (int kx = 0; kx < 3; ++kx)
        acc += w[((co * 2 + ci) * 3 + ky) * 3 + kx] *
               img[((size_t)(n * 2 + ci) * 128 + (oy + ky)) * 128 + (ox + kx)];
  float scale = gamma[co] * rsqrtf(var[co] + 1e-5f);
  float v = (acc + b[co] - mean[co]) * scale + beta[co];
  v = v > 0.f ? v : 0.f;
  out[((size_t)(n * 126 + oy) * 126 + ox) * 64 + co] = f32_to_bf16(v);
}

// ---------------------------------------------------------------------------
// conv2..8: implicit GEMM, v_wmma_f32_16x16x32_bf16
//   per wave: 16 output channels x 64 output pixels (4 accumulators), A reused
//   across 4 WMMAs; CIN templated so all load offsets are immediates.
// ---------------------------------------------------------------------------
template <int CIN>
__global__ __launch_bounds__(256) void conv3x3_wmma(
    const uint16_t* __restrict__ in,    // NHWC bf16 [N,Hin,Win,CIN]
    const uint16_t* __restrict__ wb,    // [Cout][CIN*9] bf16, BN-folded
    const float*    __restrict__ bias,  // [Cout]
    uint16_t*       __restrict__ out,   // NHWC bf16 (or NCHW-flat if nchwStore)
    int N, int Hin, int Win, int Cout, int nchwStore) {
  const int Ho = Hin - 2, Wo = Win - 2;
  const int HoWo = Ho * Wo;
  const int K = CIN * 9;
  const int coTiles = Cout >> 4;
  const int pxTiles = (HoWo + 63) >> 6;          // 64 pixels per wave
  const int tilesPerImg = coTiles * pxTiles;
  const int wave = (blockIdx.x << 3) + (threadIdx.x >> 5);
  const int lane = threadIdx.x & 31;
  const int n = wave / tilesPerImg;
  if (n >= N) return;                            // wave-uniform exit: EXEC stays full
  const int rem = wave - n * tilesPerImg;
  const int pxTile = rem / coTiles;
  const int coTile = rem - pxTile * coTiles;
  const int coBase = coTile << 4;
  const int half = lane >> 4;                    // lane group (0: lanes 0-15, 1: 16-31)
  const int idx  = lane & 15;

  // Per-lane pixel base pointers for the 4 N-tiles (clamped to stay in-bounds).
  int p[4], oyv[4], oxv[4];
  const uint16_t* bp[4];
  #pragma unroll
  for (int t = 0; t < 4; ++t) {
    p[t] = pxTile * 64 + t * 16 + idx;
    int pc = p[t] < HoWo ? p[t] : HoWo - 1;
    oyv[t] = pc / Wo; oxv[t] = pc - oyv[t] * Wo;
    bp[t] = in + ((size_t)((n * Hin + oyv[t]) * Win) + oxv[t]) * CIN + half * 16;
  }
  const uint16_t* arow = wb + (size_t)(coBase + idx) * K + half * 16;

  f32x8 acc[4];
  #pragma unroll
  for (int t = 0; t < 4; ++t) acc[t] = f32x8{0.f,0.f,0.f,0.f,0.f,0.f,0.f,0.f};

  const int rowStride = Win * CIN;               // wave-uniform
  for (int ky = 0; ky < 3; ++ky) {               // uniform pointer bumps only
    const uint16_t* ar  = arow + ky * 3 * CIN;
    const uint16_t* rb0 = bp[0] + (size_t)ky * rowStride;
    const uint16_t* rb1 = bp[1] + (size_t)ky * rowStride;
    const uint16_t* rb2 = bp[2] + (size_t)ky * rowStride;
    const uint16_t* rb3 = bp[3] + (size_t)ky * rowStride;
    #pragma unroll
    for (int kx = 0; kx < 3; ++kx) {
      #pragma unroll
      for (int cc = 0; cc < CIN; cc += 32) {
        const int o = kx * CIN + cc;             // compile-time constant offset
        ABPack ua, ub;
        ua.q[0] = *(const u32x4*)(ar + o);
        ua.q[1] = *(const u32x4*)(ar + o + 8);
        ub.q[0] = *(const u32x4*)(rb0 + o);
        ub.q[1] = *(const u32x4*)(rb0 + o + 8);
        acc[0] = wmma_bf16(ua, ub, acc[0]);
        ub.q[0] = *(const u32x4*)(rb1 + o);
        ub.q[1] = *(const u32x4*)(rb1 + o + 8);
        acc[1] = wmma_bf16(ua, ub, acc[1]);
        ub.q[0] = *(const u32x4*)(rb2 + o);
        ub.q[1] = *(const u32x4*)(rb2 + o + 8);
        acc[2] = wmma_bf16(ua, ub, acc[2]);
        ub.q[0] = *(const u32x4*)(rb3 + o);
        ub.q[1] = *(const u32x4*)(rb3 + o + 8);
        acc[3] = wmma_bf16(ua, ub, acc[3]);
      }
    }
  }

  #pragma unroll
  for (int t = 0; t < 4; ++t) {
    if (p[t] >= HoWo) continue;
    if (!nchwStore) {
      union { uint16_t h[8]; u32x4 q; } pk;
      #pragma unroll
      for (int e = 0; e < 8; ++e) {              // D: VGPR e -> co = coBase + half*8 + e
        int co = coBase + half * 8 + e;
        float f = acc[t][e] + bias[co];
        pk.h[e] = f32_to_bf16(f > 0.f ? f : 0.f);
      }
      uint16_t* op = out + ((size_t)(n * Ho + oyv[t]) * Wo + oxv[t]) * Cout
                     + coBase + half * 8;
      *(u32x4*)op = pk.q;                        // 8 contiguous channels: one b128 store
    } else {
      #pragma unroll
      for (int e = 0; e < 8; ++e) {
        int co = coBase + half * 8 + e;
        float f = acc[t][e] + bias[co];
        out[(size_t)(n * Cout + co) * HoWo + p[t]] = f32_to_bf16(f > 0.f ? f : 0.f);
      }
    }
  }
}

// ---------------------------------------------------------------------------
// 2x2 maxpool, NHWC bf16
// ---------------------------------------------------------------------------
__global__ void maxpool2_nhwc(const uint16_t* __restrict__ in, uint16_t* __restrict__ out,
                              int N, int Hi, int Wi, int C) {
  int Ho = Hi >> 1, Wo = Wi >> 1;
  long total = (long)N * Ho * Wo * C;
  long i = (long)blockIdx.x * blockDim.x + threadIdx.x;
  if (i >= total) return;
  int c = (int)(i % C); long t = i / C;
  int x = (int)(t % Wo); t /= Wo;
  int y = (int)(t % Ho); t /= Ho;
  int n = (int)t;
  const uint16_t* p0 = in + ((size_t)(n * Hi + 2 * y) * Wi + 2 * x) * C + c;
  uint16_t a = p0[0], b = p0[C];
  uint16_t d = p0[(size_t)Wi * C], e = p0[(size_t)Wi * C + C];
  uint16_t m = bf16_to_f32(a) > bf16_to_f32(b) ? a : b;
  uint16_t m2 = bf16_to_f32(d) > bf16_to_f32(e) ? d : e;
  out[i] = bf16_to_f32(m) > bf16_to_f32(m2) ? m : m2;
}

// ---------------------------------------------------------------------------
// FC1: [1024 x 8192] bf16 W  *  [8192 x 256] bf16 act  (+bias) -> f32 [256][1024]
//   per wave: 64 outputs x 16 batch; B fragment reused across 4 WMMAs.
// ---------------------------------------------------------------------------
__global__ __launch_bounds__(256) void fc1_wmma(const uint16_t* __restrict__ act,  // [256][8192]
                                                const uint16_t* __restrict__ wbf,  // [1024][8192]
                                                const float* __restrict__ fc1b,
                                                float* __restrict__ outBJ) {       // [256][1024]
  const int K = 8192;
  const int wave = (blockIdx.x << 3) + (threadIdx.x >> 5);   // 256 waves total
  if (wave >= 256) return;
  const int lane = threadIdx.x & 31;
  const int jG = wave >> 4;          // 16 groups of 4 j-tiles (64 outputs each)
  const int bTile = wave & 15;       // 16 tiles over 256 batch
  const int half = lane >> 4, idx = lane & 15;
  const uint16_t* ar[4];
  #pragma unroll
  for (int t = 0; t < 4; ++t)
    ar[t] = wbf + (size_t)((jG * 4 + t) * 16 + idx) * K + half * 16;
  const uint16_t* br = act + (size_t)(bTile * 16 + idx) * K + half * 16;

  f32x8 acc[4];
  #pragma unroll
  for (int t = 0; t < 4; ++t) acc[t] = f32x8{0.f,0.f,0.f,0.f,0.f,0.f,0.f,0.f};

  for (int kc = 0; kc < K; kc += 32) {
    ABPack ub;
    ub.q[0] = *(const u32x4*)(br + kc);
    ub.q[1] = *(const u32x4*)(br + kc + 8);
    #pragma unroll
    for (int t = 0; t < 4; ++t) {
      ABPack ua;
      ua.q[0] = *(const u32x4*)(ar[t] + kc);
      ua.q[1] = *(const u32x4*)(ar[t] + kc + 8);
      acc[t] = wmma_bf16(ua, ub, acc[t]);
    }
  }
  const int bi = bTile * 16 + idx;
  #pragma unroll
  for (int t = 0; t < 4; ++t)
    #pragma unroll
    for (int e = 0; e < 8; ++e) {
      int j = (jG * 4 + t) * 16 + half * 8 + e;
      outBJ[(size_t)bi * 1024 + j] = acc[t][e] + fc1b[j];
    }
}

// ---------------------------------------------------------------------------
// FC2 (tiny): H4pt[b][o] = dot(fc1out[b], fc2_w[o]) + fc2_b[o]
// ---------------------------------------------------------------------------
__global__ void fc2_kernel(const float* __restrict__ x, const float* __restrict__ w,
                           const float* __restrict__ b, float* __restrict__ h4) {
  int i = blockIdx.x * blockDim.x + threadIdx.x;
  if (i >= 256 * 8) return;
  int n = i >> 3, o = i & 7;
  const float* xr = x + (size_t)n * 1024;
  const float* wr = w + (size_t)o * 1024;
  float acc = b[o];
  for (int j = 0; j < 1024; ++j) acc += xr[j] * wr[j];
  h4[i] = acc;
}

// ---------------------------------------------------------------------------
// DLT: per-sample 8x8 solve (Gauss-Jordan) + 3x3 inverse -> Hinv
// ---------------------------------------------------------------------------
__global__ void dlt_hinv(const float* __restrict__ Ca, const float* __restrict__ h4,
                         float* __restrict__ hinv, int N) {
  int n = blockIdx.x * blockDim.x + threadIdx.x;
  if (n >= N) return;
  float M[8][9];
  for (int i = 0; i < 4; ++i) {
    float u = Ca[n * 8 + 2 * i], v = Ca[n * 8 + 2 * i + 1];
    float up = u + h4[n * 8 + 2 * i], vp = v + h4[n * 8 + 2 * i + 1];
    float* rA = M[2 * i];
    rA[0] = 0.f; rA[1] = 0.f; rA[2] = 0.f; rA[3] = -u; rA[4] = -v; rA[5] = -1.f;
    rA[6] = vp * u; rA[7] = vp * v; rA[8] = -vp;
    float* rB = M[2 * i + 1];
    rB[0] = u; rB[1] = v; rB[2] = 1.f; rB[3] = 0.f; rB[4] = 0.f; rB[5] = 0.f;
    rB[6] = -up * u; rB[7] = -up * v; rB[8] = up;
  }
  for (int c = 0; c < 8; ++c) {                        // partial-pivot Gauss-Jordan
    int piv = c; float best = fabsf(M[c][c]);
    for (int r = c + 1; r < 8; ++r) { float a = fabsf(M[r][c]); if (a > best) { best = a; piv = r; } }
    if (piv != c)
      for (int k = c; k < 9; ++k) { float t = M[c][k]; M[c][k] = M[piv][k]; M[piv][k] = t; }
    float inv = 1.f / M[c][c];
    for (int r = 0; r < 8; ++r) {
      if (r == c) continue;
      float f = M[r][c] * inv;
      for (int k = c; k < 9; ++k) M[r][k] -= f * M[c][k];
    }
  }
  float h[9];
  for (int c = 0; c < 8; ++c) h[c] = M[c][8] / M[c][c];
  h[8] = 1.f;
  float a = h[0], b = h[1], cc = h[2], d = h[3], e = h[4], f = h[5], g = h[6], hh = h[7], ii = h[8];
  float A0 = e * ii - f * hh, A1 = cc * hh - b * ii, A2 = b * f - cc * e;
  float A3 = f * g - d * ii, A4 = a * ii - cc * g,  A5 = cc * d - a * f;
  float A6 = d * hh - e * g, A7 = b * g - a * hh,   A8 = a * e - b * d;
  float id = 1.f / (a * A0 + b * A3 + cc * A6);
  float* o = hinv + (size_t)n * 9;
  o[0] = A0 * id; o[1] = A1 * id; o[2] = A2 * id;
  o[3] = A3 * id; o[4] = A4 * id; o[5] = A5 * id;
  o[6] = A6 * id; o[7] = A7 * id; o[8] = A8 * id;
}

// ---------------------------------------------------------------------------
// Perspective warp, zero-padded bilinear (one thread per output pixel)
// ---------------------------------------------------------------------------
DEVFN float samp(const float* img, float xi, float yi) {
  bool valid = (xi >= 0.f) && (xi <= 127.f) && (yi >= 0.f) && (yi <= 127.f);
  int xc = (int)fminf(fmaxf(xi, 0.f), 127.f);
  int yc = (int)fminf(fmaxf(yi, 0.f), 127.f);
  float v = img[yc * 128 + xc];
  return valid ? v : 0.f;
}

__global__ void warp_kernel(const float* __restrict__ Pa, const float* __restrict__ hinv,
                            float* __restrict__ out) {
  long i = (long)blockIdx.x * blockDim.x + threadIdx.x;
  if (i >= 256L * 128 * 128) return;
  int n = (int)(i >> 14);
  int r = (int)(i & 16383);
  float y = (float)(r >> 7), x = (float)(r & 127);
  const float* Hi = hinv + (size_t)n * 9;
  float xs = Hi[0] * x + Hi[1] * y + Hi[2];
  float ys = Hi[3] * x + Hi[4] * y + Hi[5];
  float wsv = Hi[6] * x + Hi[7] * y + Hi[8];
  float inv = 1.f / wsv;
  float sx = xs * inv, sy = ys * inv;
  const float* img = Pa + (size_t)n * 16384;
  float x0 = floorf(sx), y0 = floorf(sy);
  float wx = sx - x0, wy = sy - y0;
  float v = samp(img, x0, y0) * (1.f - wx) * (1.f - wy)
          + samp(img, x0 + 1.f, y0) * wx * (1.f - wy)
          + samp(img, x0, y0 + 1.f) * (1.f - wx) * wy
          + samp(img, x0 + 1.f, y0 + 1.f) * wx * wy;
  out[i] = v;
}

// ---------------------------------------------------------------------------
// Host: workspace layout + launch sequence (batch chunked x32 to live in L2)
// ---------------------------------------------------------------------------
static const size_t OFF_WB    = 0;                         // 2 MB: folded conv weights
static const size_t OFF_BIAS  = 2ull << 20;                // folded biases
static const size_t OFF_FC1W  = 4ull << 20;                // 16 MB bf16 fc1 weights
static const size_t OFF_FC1O  = 20ull << 20;               // 1 MB f32 [256][1024]
static const size_t OFF_HINV  = 21ull << 20;               // 9 KB
static const size_t OFF_ACTFC = 22ull << 20;               // 4 MB bf16 [256][8192]
static const size_t OFF_ACTA  = 32ull << 20;               // 64 MB ping
static const size_t OFF_ACTB  = 96ull << 20;               // 64 MB pong

extern "C" void kernel_launch(void* const* d_in, const int* in_sizes, int n_in,
                              void* d_out, int out_size, void* d_ws, size_t ws_size,
                              hipStream_t stream) {
  (void)in_sizes; (void)n_in; (void)out_size; (void)ws_size;
  const float* Ibatch = (const float*)d_in[0];
  const float* Ca     = (const float*)d_in[2];
  const float* Pa     = (const float*)d_in[4];
  const float* fc1w   = (const float*)d_in[53];
  const float* fc1b   = (const float*)d_in[54];
  const float* fc2w   = (const float*)d_in[55];
  const float* fc2b   = (const float*)d_in[56];

  char* ws = (char*)d_ws;
  uint16_t* WB    = (uint16_t*)(ws + OFF_WB);
  float*    BIAS  = (float*)(ws + OFF_BIAS);
  uint16_t* FC1WB = (uint16_t*)(ws + OFF_FC1W);
  float*    FC1O  = (float*)(ws + OFF_FC1O);
  float*    HINV  = (float*)(ws + OFF_HINV);
  uint16_t* ACTFC = (uint16_t*)(ws + OFF_ACTFC);
  uint16_t* ACTA  = (uint16_t*)(ws + OFF_ACTA);
  uint16_t* ACTB  = (uint16_t*)(ws + OFF_ACTB);
  float* outImg = (float*)d_out;
  float* H4     = outImg + 256 * 128 * 128;

  static const int    CinA[7]  = {64, 64, 64, 64, 128, 128, 128};     // conv2..conv8
  static const int    CoutA[7] = {64, 64, 64, 128, 128, 128, 128};
  static const size_t WBO[7]   = {0, 36864, 73728, 110592, 184320, 331776, 479232};

  // --- prep: BN-folded bf16 conv weights, bf16 fc1 weights ---
  for (int li = 0; li < 7; ++li) {
    int base = 5 + (li + 1) * 6;
    int Cin = CinA[li], Cout = CoutA[li];
    int total = Cout * Cin * 9;
    fold_weights<<<(total + 255) / 256, 256, 0, stream>>>(
        (const float*)d_in[base + 0], (const float*)d_in[base + 1],
        (const float*)d_in[base + 2], (const float*)d_in[base + 3],
        (const float*)d_in[base + 4], (const float*)d_in[base + 5],
        WB + WBO[li], BIAS + li * 128, Cin, Cout);
  }
  {
    int total = 1024 * 8192;
    cvt_bf16<<<(total + 255) / 256, 256, 0, stream>>>(fc1w, FC1WB, total);
  }

  // --- conv chain, batch chunks of 32 so activations fit in L2 ---
  const int NC = 32;
  auto conv = [&](const uint16_t* in, int li, int Hin, uint16_t* out, int nchw) {
    int Cout = CoutA[li];
    int Ho = Hin - 2; int HoWo = Ho * Ho;
    long waves = (long)NC * ((HoWo + 63) / 64) * (Cout / 16);
    unsigned grid = (unsigned)((waves + 7) / 8);
    if (CinA[li] == 64)
      conv3x3_wmma<64><<<grid, 256, 0, stream>>>(
          in, WB + WBO[li], BIAS + li * 128, out, NC, Hin, Hin, Cout, nchw);
    else
      conv3x3_wmma<128><<<grid, 256, 0, stream>>>(
          in, WB + WBO[li], BIAS + li * 128, out, NC, Hin, Hin, Cout, nchw);
  };
  auto pool = [&](const uint16_t* in, int Hi, int C, uint16_t* out) {
    int Ho = Hi / 2; long total = (long)NC * Ho * Ho * C;
    maxpool2_nhwc<<<(unsigned)((total + 255) / 256), 256, 0, stream>>>(in, out, NC, Hi, Hi, C);
  };

  for (int c = 0; c < 8; ++c) {
    const float* img = Ibatch + (size_t)c * NC * 2 * 128 * 128;
    uint16_t* actfc = ACTFC + (size_t)c * NC * 8192;
    {
      long total = (long)NC * 126 * 126 * 64;
      conv1_direct<<<(unsigned)((total + 255) / 256), 256, 0, stream>>>(
          img, (const float*)d_in[5], (const float*)d_in[6], (const float*)d_in[7],
          (const float*)d_in[8], (const float*)d_in[9], (const float*)d_in[10], ACTA, NC);
    }
    conv(ACTA, 0, 126, ACTB, 0);   // conv2 -> 124
    pool(ACTB, 124, 64, ACTA);     // -> 62
    conv(ACTA, 1, 62, ACTB, 0);    // conv3 -> 60
    conv(ACTB, 2, 60, ACTA, 0);    // conv4 -> 58
    pool(ACTA, 58, 64, ACTB);      // -> 29
    conv(ACTB, 3, 29, ACTA, 0);    // conv5 -> 27 (128ch)
    conv(ACTA, 4, 27, ACTB, 0);    // conv6 -> 25
    pool(ACTB, 25, 128, ACTA);     // -> 12
    conv(ACTA, 5, 12, ACTB, 0);    // conv7 -> 10
    conv(ACTB, 6, 10, actfc, 1);   // conv8 -> 8, NCHW-flat bf16 for FC
  }

  // --- heads ---
  fc1_wmma<<<32, 256, 0, stream>>>(ACTFC, FC1WB, fc1b, FC1O);
  fc2_kernel<<<8, 256, 0, stream>>>(FC1O, fc2w, fc2b, H4);
  dlt_hinv<<<1, 256, 0, stream>>>(Ca, H4, HINV, 256);
  {
    long total = 256L * 128 * 128;
    warp_kernel<<<(unsigned)((total + 255) / 256), 256, 0, stream>>>(Pa, HINV, outImg);
  }
}
